// TransformerConvLayer_80668075753644
// MI455X (gfx1250) — matrix operand order repaired and checked by reading, not compiled, not run on hardware
//
#include <hip/hip_runtime.h>

// ---------------------------------------------------------------------------
// GraphGPS-style TransformerConv layer for MI455X (gfx1250, wave32, WMMA).
// GEMMs: f16 inputs, v_wmma_f32_16x16x32_f16, fp32 accumulate.
// Residual-add + BatchNorm statistics fused into the GEMM epilogue to cut
// HBM traffic on the edge path (memory-bound at 23.3 TB/s).
// ---------------------------------------------------------------------------

namespace {

constexpr int kN = 40000;   // nodes
constexpr int kE = 640000;  // edges
constexpr int kD = 128;     // model dim
constexpr int kH = 8;       // heads
constexpr long long kND = (long long)kN * kD;
constexpr long long kED = (long long)kE * kD;

typedef _Float16 h8   __attribute__((ext_vector_type(8)));
typedef _Float16 v16h __attribute__((ext_vector_type(16)));
typedef float    v8f  __attribute__((ext_vector_type(8)));

// ------------------------------ utility kernels ----------------------------

__global__ void fill_kernel(float* __restrict__ p, float v, long long n) {
  long long i = (long long)blockIdx.x * blockDim.x + threadIdx.x;
  if (i < n) p[i] = v;
}

__global__ void conv_h16(const float* __restrict__ in, _Float16* __restrict__ out,
                         long long n) {
  long long i = (long long)blockIdx.x * blockDim.x + threadIdx.x;
  if (i < n) out[i] = (_Float16)in[i];
}

// in: [K, NC] row-major f32  ->  out: [NC, K] row-major f16 (i.e. B transposed)
__global__ void conv_T16(const float* __restrict__ in, _Float16* __restrict__ out,
                         int K, int NC) {
  int i = blockIdx.x * blockDim.x + threadIdx.x;
  if (i >= K * NC) return;
  int n = i / K, k = i - n * K;
  out[i] = (_Float16)in[k * NC + n];
}

// ------------------------------ WMMA GEMM ----------------------------------
// C[M,NC] = A[M,K](f16) * B[K,NC] (+bias) (+resid) (relu?)  (Bt[NC,K] f16)
// Optional fused BatchNorm statistics: per-column sum/sumsq of the written
// fp32 result, reduced in LDS then flushed with global atomics (NC==128 only).
// Block = 256 threads = 8 wave32 waves; each wave computes 16 rows x NC cols.
template <int NC>
__global__ __launch_bounds__(256) void gemm_wmma(
    const _Float16* __restrict__ A, const _Float16* __restrict__ Bt,
    const float* __restrict__ bias, const float* __restrict__ resid,
    float* __restrict__ Cf, _Float16* __restrict__ Ch,
    float* __restrict__ s_sum, float* __restrict__ s_sumsq,
    int M, int K, int relu) {
  extern __shared__ _Float16 lds_b[];  // NC x 32 f16 (one K-chunk of Bt)
  __shared__ float ls[kD], lsq[kD];
  constexpr int NT = NC / 16;
  const int tid = threadIdx.x;
  const int lane = tid & 31;
  const int wave = tid >> 5;
  const int lrow = lane & 15;
  const int half = lane >> 4;
  const int rowbase = blockIdx.x * 128 + wave * 16;
  int arow = rowbase + lrow;
  if (arow >= M) arow = M - 1;  // clamp loads; stores guarded below

  const bool do_stats = (s_sum != nullptr);
  if (do_stats && tid < kD) { ls[tid] = 0.0f; lsq[tid] = 0.0f; }

  v8f acc[NT];
#pragma unroll
  for (int t = 0; t < NT; ++t)
#pragma unroll
    for (int r = 0; r < 8; ++r) acc[t][r] = 0.0f;

  const int KC = K >> 5;
  for (int kc = 0; kc < KC; ++kc) {
    // stage Bt[:, kc*32 .. kc*32+31] into LDS (contiguous 64B per column row)
    for (int idx = tid; idx < NC * 4; idx += 256) {
      int n = idx >> 2, j8 = (idx & 3) << 3;
      *(h8*)(lds_b + n * 32 + j8) =
          *(const h8*)(Bt + (long long)n * K + kc * 32 + j8);
    }
    __syncthreads();

    // A fragment: lane holds row (lane&15); K = kc*32 + {half*8..+7, 16+half*8..+7}
    const _Float16* Ar = A + (long long)arow * K + kc * 32 + half * 8;
    h8 alo = *(const h8*)(Ar);
    h8 ahi = *(const h8*)(Ar + 16);
    if (kc + 1 < KC) __builtin_prefetch(Ar + 32, 0, 3);  // next K-chunk of A
    v16h a;
#pragma unroll
    for (int i = 0; i < 8; ++i) { a[i] = alo[i]; a[i + 8] = ahi[i]; }

#pragma unroll
    for (int t = 0; t < NT; ++t) {
      // B fragment: lane holds col (t*16 + lane&15); K = {half*8..+7, 16+half*8..+7}
      const _Float16* Bp = lds_b + (t * 16 + lrow) * 32 + half * 8;
      h8 blo = *(const h8*)(Bp);
      h8 bhi = *(const h8*)(Bp + 16);
      v16h b;
#pragma unroll
      for (int i = 0; i < 8; ++i) { b[i] = blo[i]; b[i + 8] = bhi[i]; }
      acc[t] = __builtin_amdgcn_wmma_f32_16x16x32_f16(
          false, a, false, b, (short)0, acc[t], false, false);
    }
    __syncthreads();
  }

  // Epilogue: C layout — VGPR r: M = r + 8*half; N = lane&15
#pragma unroll
  for (int t = 0; t < NT; ++t) {
    int col = t * 16 + lrow;
    float bv = bias ? bias[col] : 0.0f;
#pragma unroll
    for (int r = 0; r < 8; ++r) {
      int row = rowbase + r + half * 8;
      if (row < M) {
        long long o = (long long)row * NC + col;
        float v = acc[t][r] + bv;
        if (resid) v += resid[o];
        if (relu && v < 0.0f) v = 0.0f;
        if (Cf) Cf[o] = v;
        if (Ch) Ch[o] = (_Float16)v;
        if (do_stats) {
          atomicAdd(&ls[col & (kD - 1)], v);
          atomicAdd(&lsq[col & (kD - 1)], v * v);
        }
      }
    }
  }
  if (do_stats) {
    __syncthreads();
    if (tid < kD) {
      atomicAdd(&s_sum[tid], ls[tid]);
      atomicAdd(&s_sumsq[tid], lsq[tid]);
    }
  }
}

// ------------------------------ attention ----------------------------------

__device__ __forceinline__ void atomicMaxF(float* addr, float val) {
  int* ia = (int*)addr;
  int old = __float_as_int(*addr);
  while (__int_as_float(old) < val) {
    int assumed = old;
    old = atomicCAS(ia, assumed, __float_as_int(val));
    if (old == assumed) break;
  }
}

// pass 1: score = q[dst] . (k[src] + ee) / sqrt(16); segment max
__global__ void attn_score_kernel(const float* __restrict__ q,
                                  const float* __restrict__ k,
                                  const float* __restrict__ ee,
                                  const int* __restrict__ src,
                                  const int* __restrict__ dst,
                                  float* __restrict__ score,
                                  float* __restrict__ mbuf) {
  long long i = (long long)blockIdx.x * blockDim.x + threadIdx.x;
  if (i >= (long long)kE * kH) return;
  int e = (int)(i >> 3), h = (int)(i & 7);
  int s = src[e], d = dst[e];
  const float4* qp = (const float4*)(q + (long long)d * kD + h * 16);
  const float4* kp = (const float4*)(k + (long long)s * kD + h * 16);
  const float4* ep = (const float4*)(ee + (long long)e * kD + h * 16);
  float acc = 0.0f;
#pragma unroll
  for (int j = 0; j < 4; ++j) {
    float4 a = qp[j], b = kp[j], c = ep[j];
    acc += a.x * (b.x + c.x) + a.y * (b.y + c.y) + a.z * (b.z + c.z) +
           a.w * (b.w + c.w);
  }
  acc *= 0.25f;  // 1/sqrt(DH)
  score[i] = acc;
  atomicMaxF(mbuf + (long long)d * kH + h, acc);
}

// pass 2: p = exp(score - m[dst]); segment sum
__global__ void attn_exp_kernel(float* __restrict__ score,
                                const float* __restrict__ mbuf,
                                float* __restrict__ ssum,
                                const int* __restrict__ dst) {
  long long i = (long long)blockIdx.x * blockDim.x + threadIdx.x;
  if (i >= (long long)kE * kH) return;
  int e = (int)(i >> 3), h = (int)(i & 7);
  int d = dst[e];
  float p = expf(score[i] - mbuf[(long long)d * kH + h]);
  score[i] = p;
  atomicAdd(ssum + (long long)d * kH + h, p);
}

// pass 3: agg[dst] += alpha * (v[src] + ee)   (agg pre-seeded with x@Wskip)
__global__ void attn_scatter_kernel(const float* __restrict__ p,
                                    const float* __restrict__ ssum,
                                    const float* __restrict__ v,
                                    const float* __restrict__ ee,
                                    const int* __restrict__ src,
                                    const int* __restrict__ dst,
                                    float* __restrict__ agg) {
  long long i = (long long)blockIdx.x * blockDim.x + threadIdx.x;
  if (i >= (long long)kE * kH) return;
  int e = (int)(i >> 3), h = (int)(i & 7);
  int s = src[e], d = dst[e];
  float alpha = p[i] / (ssum[(long long)d * kH + h] + 1e-16f);
  const float* vp = v + (long long)s * kD + h * 16;
  const float* ep = ee + (long long)e * kD + h * 16;
  float* ap = agg + (long long)d * kD + h * 16;
#pragma unroll
  for (int j = 0; j < 16; ++j) atomicAdd(ap + j, alpha * (vp[j] + ep[j]));
}

// ------------------------------ batchnorm apply ----------------------------

// y = (res - mean) * rsqrt(var+eps) * g + b; optional f32 and f16 outputs
__global__ void bn_apply_kernel(const float* __restrict__ res,
                                const float* __restrict__ sum,
                                const float* __restrict__ sumsq,
                                const float* __restrict__ g,
                                const float* __restrict__ b,
                                float* __restrict__ outf,
                                _Float16* __restrict__ outh, long long M) {
  long long i = (long long)blockIdx.x * blockDim.x + threadIdx.x;
  if (i >= M * kD) return;
  int col = (int)(i & (kD - 1));
  float inv = 1.0f / (float)M;
  float mean = sum[col] * inv;
  float var = sumsq[col] * inv - mean * mean;
  float sc = rsqrtf(var + 1e-5f) * g[col];
  float v = (res[i] - mean) * sc + b[col];
  if (outf) outf[i] = v;
  if (outh) outh[i] = (_Float16)v;
}

inline int gridFor(long long n, int blk) { return (int)((n + blk - 1) / blk); }

}  // namespace

// ------------------------------- launcher ----------------------------------

extern "C" void kernel_launch(void* const* d_in, const int* in_sizes, int n_in,
                              void* d_out, int out_size, void* d_ws,
                              size_t ws_size, hipStream_t stream) {
  (void)in_sizes; (void)n_in; (void)out_size; (void)ws_size;
  // ---- inputs (setup_inputs dict order) ----
  const float* x         = (const float*)d_in[0];
  const float* edge_attr = (const float*)d_in[1];
  const int*   eidx      = (const int*)d_in[2];
  const float* Wq = (const float*)d_in[3];
  const float* Wk = (const float*)d_in[4];
  const float* Wv = (const float*)d_in[5];
  const float* We = (const float*)d_in[6];
  const float* Wskip = (const float*)d_in[7];
  const float* OhW = (const float*)d_in[8];
  const float* Ohb = (const float*)d_in[9];
  const float* OeW = (const float*)d_in[10];
  const float* Oeb = (const float*)d_in[11];
  const float* bn1h_g = (const float*)d_in[12];
  const float* bn1h_b = (const float*)d_in[13];
  const float* bn1e_g = (const float*)d_in[14];
  const float* bn1e_b = (const float*)d_in[15];
  const float* bn2h_g = (const float*)d_in[16];
  const float* bn2h_b = (const float*)d_in[17];
  const float* bn2e_g = (const float*)d_in[18];
  const float* bn2e_b = (const float*)d_in[19];
  const float* FhW1 = (const float*)d_in[20];
  const float* Fhb1 = (const float*)d_in[21];
  const float* FhW2 = (const float*)d_in[22];
  const float* Fhb2 = (const float*)d_in[23];
  const float* FeW1 = (const float*)d_in[24];
  const float* Feb1 = (const float*)d_in[25];
  const float* FeW2 = (const float*)d_in[26];
  const float* Feb2 = (const float*)d_in[27];

  const int* src = eidx;        // edge_index[0]
  const int* dst = eidx + kE;   // edge_index[1]

  float* out_h = (float*)d_out;
  float* out_e = out_h + kND;

  // ---- workspace layout ----
  char* w = (char*)d_ws;
  size_t off = 0;
  auto take = [&](size_t bytes) -> void* {
    void* p = w + off;
    off = (off + bytes + 255) & ~(size_t)255;
    return p;
  };
  _Float16* WqT    = (_Float16*)take(kD * kD * 2);
  _Float16* WkT    = (_Float16*)take(kD * kD * 2);
  _Float16* WvT    = (_Float16*)take(kD * kD * 2);
  _Float16* WeT    = (_Float16*)take(kD * kD * 2);
  _Float16* WskipT = (_Float16*)take(kD * kD * 2);
  _Float16* OhT    = (_Float16*)take(kD * kD * 2);
  _Float16* OeT    = (_Float16*)take(kD * kD * 2);
  _Float16* Fh1T   = (_Float16*)take(kD * 2 * kD * 2);  // [256,128]
  _Float16* Fh2T   = (_Float16*)take(kD * 2 * kD * 2);  // [128,256]
  _Float16* Fe1T   = (_Float16*)take(kD * 2 * kD * 2);
  _Float16* Fe2T   = (_Float16*)take(kD * 2 * kD * 2);
  float* stats = (float*)take(4 * 256 * sizeof(float));  // 4 BN x (sum|sumsq)
  float* st_h1 = stats + 0,   *st_h1q = stats + 128;
  float* st_e1 = stats + 256, *st_e1q = stats + 384;
  float* st_h2 = stats + 512, *st_h2q = stats + 640;
  float* st_e2 = stats + 768, *st_e2q = stats + 896;
  float* mbuf  = (float*)take((long long)kN * kH * 4);
  float* ssum  = (float*)take((long long)kN * kH * 4);
  float* score = (float*)take((long long)kE * kH * 4);
  _Float16* xb = (_Float16*)take(kND * 2);
  _Float16* hb = (_Float16*)take(kND * 2);
  float* q    = (float*)take(kND * 4);
  float* kbuf = (float*)take(kND * 4);
  float* vbuf = (float*)take(kND * 4);
  float* agg  = (float*)take(kND * 4);
  _Float16* ht256 = (_Float16*)take((long long)kN * 2 * kD * 2);
  _Float16* eb = (_Float16*)take(kED * 2);
  float* ee   = (float*)take(kED * 4);   // later reused as et256 (E x 256 f16)
  float* ebn  = (float*)take(kED * 4);   // BN1e fp32 output (kept for residual)
  _Float16* et256 = (_Float16*)ee;       // alias: same byte size

  const int BLK = 256;
  // ---- 1) weight transpose + f16 convert ----
  int wg128 = gridFor(kD * kD, BLK), wg256 = gridFor(kD * 2 * kD, BLK);
  conv_T16<<<wg128, BLK, 0, stream>>>(Wq, WqT, kD, kD);
  conv_T16<<<wg128, BLK, 0, stream>>>(Wk, WkT, kD, kD);
  conv_T16<<<wg128, BLK, 0, stream>>>(Wv, WvT, kD, kD);
  conv_T16<<<wg128, BLK, 0, stream>>>(We, WeT, kD, kD);
  conv_T16<<<wg128, BLK, 0, stream>>>(Wskip, WskipT, kD, kD);
  conv_T16<<<wg128, BLK, 0, stream>>>(OhW, OhT, kD, kD);
  conv_T16<<<wg128, BLK, 0, stream>>>(OeW, OeT, kD, kD);
  conv_T16<<<wg256, BLK, 0, stream>>>(FhW1, Fh1T, kD, 2 * kD);       // [256,128]
  conv_T16<<<wg256, BLK, 0, stream>>>(FhW2, Fh2T, 2 * kD, kD);       // [128,256]
  conv_T16<<<wg256, BLK, 0, stream>>>(FeW1, Fe1T, kD, 2 * kD);
  conv_T16<<<wg256, BLK, 0, stream>>>(FeW2, Fe2T, 2 * kD, kD);

  // ---- 2) activation converts + init ----
  conv_h16<<<gridFor(kND, BLK), BLK, 0, stream>>>(x, xb, kND);
  conv_h16<<<gridFor(kED, BLK), BLK, 0, stream>>>(edge_attr, eb, kED);
  fill_kernel<<<gridFor((long long)kN * kH, BLK), BLK, 0, stream>>>(
      mbuf, -INFINITY, (long long)kN * kH);
  fill_kernel<<<gridFor((long long)kN * kH, BLK), BLK, 0, stream>>>(
      ssum, 0.0f, (long long)kN * kH);
  fill_kernel<<<4, BLK, 0, stream>>>(stats, 0.0f, 1024);

  // ---- 3) projection GEMMs (WMMA) ----
  int gN = gridFor(kN, 128), gEg = gridFor(kE, 128);
  size_t sh128 = 128 * 32 * 2, sh256 = 256 * 32 * 2;
  gemm_wmma<128><<<gN, BLK, sh128, stream>>>(xb, WqT, nullptr, nullptr, q, nullptr,
                                             nullptr, nullptr, kN, kD, 0);
  gemm_wmma<128><<<gN, BLK, sh128, stream>>>(xb, WkT, nullptr, nullptr, kbuf, nullptr,
                                             nullptr, nullptr, kN, kD, 0);
  gemm_wmma<128><<<gN, BLK, sh128, stream>>>(xb, WvT, nullptr, nullptr, vbuf, nullptr,
                                             nullptr, nullptr, kN, kD, 0);
  gemm_wmma<128><<<gN, BLK, sh128, stream>>>(xb, WskipT, nullptr, nullptr, agg, nullptr,
                                             nullptr, nullptr, kN, kD, 0);  // seed agg
  gemm_wmma<128><<<gEg, BLK, sh128, stream>>>(eb, WeT, nullptr, nullptr, ee, nullptr,
                                              nullptr, nullptr, kE, kD, 0);
  // Oe GEMM fused with residual (edge_attr) + BN1e statistics -> out_e
  gemm_wmma<128><<<gEg, BLK, sh128, stream>>>(eb, OeT, Oeb, edge_attr, out_e, nullptr,
                                              st_e1, st_e1q, kE, kD, 0);

  // ---- 4) BN1e apply: out_e -> ebn (f32 for final residual) + eb (f16 for FFN)
  bn_apply_kernel<<<gridFor(kED, BLK), BLK, 0, stream>>>(
      out_e, st_e1, st_e1q, bn1e_g, bn1e_b, ebn, eb, kE);

  // ---- 5) attention (segment softmax + scatter) ----
  long long EH = (long long)kE * kH;
  attn_score_kernel<<<gridFor(EH, BLK), BLK, 0, stream>>>(q, kbuf, ee, src, dst, score, mbuf);
  attn_exp_kernel<<<gridFor(EH, BLK), BLK, 0, stream>>>(score, mbuf, ssum, dst);
  attn_scatter_kernel<<<gridFor(EH, BLK), BLK, 0, stream>>>(score, ssum, vbuf, ee, src, dst, agg);

  // ---- 6) node O_h GEMM fused with residual (x) + BN1h stats -> out_h ----
  conv_h16<<<gridFor(kND, BLK), BLK, 0, stream>>>(agg, hb, kND);
  gemm_wmma<128><<<gN, BLK, sh128, stream>>>(hb, OhT, Ohb, x, out_h, nullptr,
                                             st_h1, st_h1q, kN, kD, 0);
  bn_apply_kernel<<<gridFor(kND, BLK), BLK, 0, stream>>>(
      out_h, st_h1, st_h1q, bn1h_g, bn1h_b, agg /*h_bn f32*/, hb /*h_bn f16*/, kN);

  // ---- 7) node FFN; second GEMM fused with residual (h_bn) + BN2h stats ----
  gemm_wmma<256><<<gN, BLK, sh256, stream>>>(hb, Fh1T, Fhb1, nullptr, nullptr, ht256,
                                             nullptr, nullptr, kN, kD, 1);
  gemm_wmma<128><<<gN, BLK, sh128, stream>>>(ht256, Fh2T, Fhb2, agg, out_h, nullptr,
                                             st_h2, st_h2q, kN, 2 * kD, 0);
  bn_apply_kernel<<<gridFor(kND, BLK), BLK, 0, stream>>>(
      out_h, st_h2, st_h2q, bn2h_g, bn2h_b, out_h, nullptr, kN);

  // ---- 8) edge FFN (reuses ee region; attention finished); second GEMM fused
  //          with residual (e_bn) + BN2e stats -> out_e ----
  gemm_wmma<256><<<gEg, BLK, sh256, stream>>>(eb, Fe1T, Feb1, nullptr, nullptr, et256,
                                              nullptr, nullptr, kE, kD, 1);
  gemm_wmma<128><<<gEg, BLK, sh128, stream>>>(et256, Fe2T, Feb2, ebn, out_e, nullptr,
                                              st_e2, st_e2q, kE, 2 * kD, 0);
  bn_apply_kernel<<<gridFor(kED, BLK), BLK, 0, stream>>>(
      out_e, st_e2, st_e2q, bn2e_g, bn2e_b, out_e, nullptr, kE);
}
